// BertSelfAttention_23811298689566
// MI455X (gfx1250) — matrix operand order repaired and checked by reading, not compiled, not run on hardware
//
#include <hip/hip_runtime.h>

#define BB 4
#define SS 2048
#define DD 1024
#define HH 16
#define HD 64

typedef __attribute__((ext_vector_type(4)))  float          v4f;
typedef __attribute__((ext_vector_type(8)))  float          v8f;
typedef __attribute__((ext_vector_type(8)))  unsigned short us8;
typedef __attribute__((ext_vector_type(16))) unsigned short us16;
typedef __attribute__((ext_vector_type(16))) __bf16         v16bf;

#define LOG2E 1.4426950408889634f

__device__ __forceinline__ unsigned short f2bf(float f) {
  unsigned u = __builtin_bit_cast(unsigned, f);
  u += 0x7FFFu + ((u >> 16) & 1u);   // round-to-nearest-even
  return (unsigned short)(u >> 16);
}

__device__ __forceinline__ float fast_exp2(float x) {
#if __has_builtin(__builtin_amdgcn_exp2f)
  return __builtin_amdgcn_exp2f(x);   // raw v_exp_f32
#else
  return exp2f(x);
#endif
}

__device__ __forceinline__ v8f wmma_bf16(us16 a, us16 b, v8f c) {
  return __builtin_amdgcn_wmma_f32_16x16x32_bf16(
      false, __builtin_bit_cast(v16bf, a),
      false, __builtin_bit_cast(v16bf, b),
      (short)0, c, false, false);
}

__device__ __forceinline__ float redmax16(float v) {
  v = fmaxf(v, __shfl_xor(v, 1, 32));
  v = fmaxf(v, __shfl_xor(v, 2, 32));
  v = fmaxf(v, __shfl_xor(v, 4, 32));
  v = fmaxf(v, __shfl_xor(v, 8, 32));
  return v;
}
__device__ __forceinline__ float redsum16(float v) {
  v += __shfl_xor(v, 1, 32);
  v += __shfl_xor(v, 2, 32);
  v += __shfl_xor(v, 4, 32);
  v += __shfl_xor(v, 8, 32);
  return v;
}

// ---------------------------------------------------------------------------
// Kernel 1: QKV projection.  C[m,n] = X[m,:] @ W[:,n] + bias[n]  (bf16 out)
// grid = (M/64, N/64, 3), block = 128 (4 waves).
// z: 0=Q (scaled by 0.125*log2e) 1=K 2=V (stored transposed [bh][hd][s])
// ---------------------------------------------------------------------------
__global__ __launch_bounds__(128) void qkv_proj(
    const float* __restrict__ X,
    const float* __restrict__ Wq, const float* __restrict__ Wk,
    const float* __restrict__ Wv,
    const float* __restrict__ bq, const float* __restrict__ bk,
    const float* __restrict__ bv,
    unsigned short* __restrict__ qws, unsigned short* __restrict__ kws,
    unsigned short* __restrict__ vtws)
{
  __shared__ alignas(16) unsigned short lA[64 * 40];  // [row][k], stride 40
  __shared__ alignas(16) unsigned short lB[64 * 40];  // [n][k],   stride 40

  const int z = blockIdx.z;
  const float* W    = (z == 0) ? Wq : (z == 1) ? Wk : Wv;
  const float* bias = (z == 0) ? bq : (z == 1) ? bk : bv;
  unsigned short* outp = (z == 0) ? qws : (z == 1) ? kws : vtws;
  const float scale = (z == 0) ? 0.125f * LOG2E : 1.0f;
  const bool  vt    = (z == 2);

  const int m0 = blockIdx.x * 64;
  const int n0 = blockIdx.y * 64;
  const int t  = threadIdx.x;
  const int lane = t & 31, wid = t >> 5;
  const int c  = lane & 15;        // col-in-tile / row-in-A-frag
  const int hf = lane >> 4;        // lane half
  const int kb = hf * 8;           // unused (A-frag uses bk16)
  const int bk16 = hf * 16;        // B-frag k base

  v8f acc0 = {}, acc1 = {}, acc2 = {}, acc3 = {};

  for (int k0 = 0; k0 < DD; k0 += 32) {
    // ---- stage X tile (64 rows x 32 k) -> bf16 LDS, row-major ----
    {
      const int row = t >> 1, half = t & 1;
      const float* src = X + (size_t)(m0 + row) * DD + k0 + half * 16;
      v4f a0 = ((const v4f*)src)[0];
      v4f a1 = ((const v4f*)src)[1];
      v4f a2 = ((const v4f*)src)[2];
      v4f a3 = ((const v4f*)src)[3];
      us8 lo, hi;
#pragma unroll
      for (int i = 0; i < 4; ++i) {
        lo[i] = f2bf(a0[i]); lo[4 + i] = f2bf(a1[i]);
        hi[i] = f2bf(a2[i]); hi[4 + i] = f2bf(a3[i]);
      }
      us8* dst = (us8*)&lA[row * 40 + half * 16];
      dst[0] = lo;
      dst[1] = hi;
    }
    // ---- stage W tile (32 k x 64 n) -> bf16 LDS, transposed [n][k] ----
    {
      const int kk = t >> 2, qq = t & 3;
      const float* src = W + (size_t)(k0 + kk) * DD + n0 + qq * 16;
#pragma unroll
      for (int j = 0; j < 16; j += 4) {
        v4f x = *(const v4f*)(src + j);
#pragma unroll
        for (int i = 0; i < 4; ++i)
          lB[(qq * 16 + j + i) * 40 + kk] = f2bf(x[i]);
      }
    }
    __syncthreads();
    // ---- WMMA: preload all fragments, then 4 back-to-back WMMAs ----
    {
      const unsigned short* ap = &lA[(wid * 16 + c) * 40];
      us8 lo = *(const us8*)(ap + bk16);
      us8 hi = *(const us8*)(ap + bk16 + 8);
      us16 afrag;
#pragma unroll
      for (int i = 0; i < 8; ++i) { afrag[i] = lo[i]; afrag[8 + i] = hi[i]; }

      us16 bf0 = *(const us16*)(&lB[(0 * 16 + c) * 40 + bk16]);
      us16 bf1 = *(const us16*)(&lB[(1 * 16 + c) * 40 + bk16]);
      us16 bf2 = *(const us16*)(&lB[(2 * 16 + c) * 40 + bk16]);
      us16 bf3 = *(const us16*)(&lB[(3 * 16 + c) * 40 + bk16]);
      acc0 = wmma_bf16(afrag, bf0, acc0);
      acc1 = wmma_bf16(afrag, bf1, acc1);
      acc2 = wmma_bf16(afrag, bf2, acc2);
      acc3 = wmma_bf16(afrag, bf3, acc3);
    }
    __syncthreads();
  }

  // ---- epilogue: bias, scale, bf16 store ----
  auto store_tile = [&](v8f a, int nt) {
    const int n = n0 + nt * 16 + c;
    const float bsv = bias[n];
    const int h = n >> 6, hd = n & 63;
    const int mbase = m0 + wid * 16 + hf * 8;
    const int b = mbase >> 11;
    const int s = mbase & (SS - 1);
    if (vt) {
      // V^T: rows are contiguous in s -> one 16-byte store
      us8 pk;
#pragma unroll
      for (int r = 0; r < 8; ++r) pk[r] = f2bf(a[r] + bsv);
      *(us8*)&outp[((size_t)(b * HH + h) * HD + hd) * SS + s] = pk;
    } else {
#pragma unroll
      for (int r = 0; r < 8; ++r) {
        const float val = (a[r] + bsv) * scale;
        outp[((size_t)(b * HH + h) * SS + (s + r)) * HD + hd] = f2bf(val);
      }
    }
  };
  store_tile(acc0, 0);
  store_tile(acc1, 1);
  store_tile(acc2, 2);
  store_tile(acc3, 3);
}

// ---------------------------------------------------------------------------
// Kernel 2: flash attention.  One wave per 16-row Q tile, 64-key blocks.
// grid = B*H*(S/16)/4 blocks, block = 128 (4 independent waves).
// ---------------------------------------------------------------------------
__global__ __launch_bounds__(128) void flash_attn(
    const unsigned short* __restrict__ qws,
    const unsigned short* __restrict__ kws,
    const unsigned short* __restrict__ vtws,
    const float* __restrict__ mask,
    float* __restrict__ out)
{
  __shared__ alignas(16) unsigned short lP[4][16 * 72];  // per-wave P staging

  const int lane = threadIdx.x & 31;
  const int wid  = threadIdx.x >> 5;
  const int wgid = blockIdx.x * 4 + wid;
  const int bh = wgid >> 7;            // 128 q-tiles per (b,h)
  const int q0 = (wgid & 127) * 16;
  const int b  = bh >> 4;
  const int h  = bh & 15;

  const int c  = lane & 15;
  const int hf = lane >> 4;
  const int kb = hf * 8;

  // Q A-fragments (16 x 64, two 16x32 k-steps), resident in VGPRs
  us16 qa[2];
  {
    const unsigned short* qp = qws + ((size_t)bh * SS + q0 + c) * HD;
#pragma unroll
    for (int ks = 0; ks < 2; ++ks) {
      us8 lo = *(const us8*)(qp + ks * 32 + hf * 16);
      us8 hi = *(const us8*)(qp + ks * 32 + hf * 16 + 8);
#pragma unroll
      for (int i = 0; i < 8; ++i) { qa[ks][i] = lo[i]; qa[ks][8 + i] = hi[i]; }
    }
  }

  v8f o0 = {}, o1 = {}, o2 = {}, o3 = {};   // O tile 16x64, f32
  float run_m[8], run_l[8];                 // run_l = per-lane partial sums
#pragma unroll
  for (int r = 0; r < 8; ++r) { run_m[r] = -3.4e38f; run_l[r] = 0.f; }

  const float* mb = mask + (size_t)b * SS;
  unsigned short* myP = lP[wid];

  for (int key0 = 0; key0 < SS; key0 += 64) {
    if (key0 + 64 < SS) {   // prefetch next K/V key block (L2-resident stream)
      __builtin_prefetch(kws + ((size_t)bh * SS + key0 + 64 + lane) * HD, 0, 1);
      __builtin_prefetch(vtws + ((size_t)bh * HD + lane) * SS + key0 + 64, 0, 1);
    }
    // ---- scores (exp2 domain): four 16x16 tiles over 64 keys ----
    v8f s0 = {}, s1 = {}, s2 = {}, s3 = {};
#pragma unroll
    for (int ks = 0; ks < 2; ++ks) {
      const unsigned short* kp =
          kws + ((size_t)bh * SS + key0 + c) * HD + ks * 32 + hf * 16;
      us16 b0 = *(const us16*)(kp);
      us16 b1 = *(const us16*)(kp + 16 * HD);
      us16 b2 = *(const us16*)(kp + 32 * HD);
      us16 b3 = *(const us16*)(kp + 48 * HD);
      s0 = wmma_bf16(qa[ks], b0, s0);
      s1 = wmma_bf16(qa[ks], b1, s1);
      s2 = wmma_bf16(qa[ks], b2, s2);
      s3 = wmma_bf16(qa[ks], b3, s3);
    }
    // ---- mask add + online softmax (exp2 domain) ----
    const float mk0 = mb[key0 + c]      * LOG2E;
    const float mk1 = mb[key0 + 16 + c] * LOG2E;
    const float mk2 = mb[key0 + 32 + c] * LOG2E;
    const float mk3 = mb[key0 + 48 + c] * LOG2E;
#pragma unroll
    for (int r = 0; r < 8; ++r) {
      s0[r] += mk0; s1[r] += mk1; s2[r] += mk2; s3[r] += mk3;
      float bm = fmaxf(fmaxf(s0[r], s1[r]), fmaxf(s2[r], s3[r]));
      bm = redmax16(bm);
      const float nm = fmaxf(run_m[r], bm);
      const float alpha = fast_exp2(run_m[r] - nm);
      run_m[r] = nm;
      const float p0 = fast_exp2(s0[r] - nm);
      const float p1 = fast_exp2(s1[r] - nm);
      const float p2 = fast_exp2(s2[r] - nm);
      const float p3 = fast_exp2(s3[r] - nm);
      s0[r] = p0; s1[r] = p1; s2[r] = p2; s3[r] = p3;
      run_l[r] = run_l[r] * alpha + ((p0 + p1) + (p2 + p3));  // per-lane partial
      o0[r] *= alpha; o1[r] *= alpha; o2[r] *= alpha; o3[r] *= alpha;
    }
    // ---- C-layout P (f32) -> LDS bf16 (16x64, row stride 72) ----
#pragma unroll
    for (int r = 0; r < 8; ++r) {
      unsigned short* pr = &myP[(hf * 8 + r) * 72 + c];
      pr[0]  = f2bf(s0[r]);
      pr[16] = f2bf(s1[r]);
      pr[32] = f2bf(s2[r]);
      pr[48] = f2bf(s3[r]);
    }
    __syncthreads();
    // ---- re-read P in A-fragment layout (two 16x32 k-steps) ----
    us16 pa0, pa1;
    {
      const unsigned short* pp = &myP[c * 72];
      us8 l0 = *(const us8*)(pp + hf * 16);
      us8 h0 = *(const us8*)(pp + hf * 16 + 8);
      us8 l1 = *(const us8*)(pp + 32 + hf * 16);
      us8 h1 = *(const us8*)(pp + 40 + hf * 16);
#pragma unroll
      for (int i = 0; i < 8; ++i) {
        pa0[i] = l0[i]; pa0[8 + i] = h0[i];
        pa1[i] = l1[i]; pa1[8 + i] = h1[i];
      }
    }
    // ---- O += P @ V  (V stored transposed [bh][hd][s]) ----
#pragma unroll
    for (int ks = 0; ks < 2; ++ks) {
      const us16 pa = ks ? pa1 : pa0;
      const unsigned short* vp =
          vtws + ((size_t)bh * HD + c) * SS + key0 + ks * 32 + hf * 16;
      us16 v0 = *(const us16*)(vp);
      us16 v1 = *(const us16*)(vp + 16 * SS);
      us16 v2 = *(const us16*)(vp + 32 * SS);
      us16 v3 = *(const us16*)(vp + 48 * SS);
      o0 = wmma_bf16(pa, v0, o0);
      o1 = wmma_bf16(pa, v1, o1);
      o2 = wmma_bf16(pa, v2, o2);
      o3 = wmma_bf16(pa, v3, o3);
    }
    __syncthreads();
  }

  // ---- single end-of-kernel cross-lane reduction of the softmax sums ----
#pragma unroll
  for (int r = 0; r < 8; ++r) run_l[r] = redsum16(run_l[r]);

  // ---- normalize and store fp32 output [B,S,D] ----
  const size_t orow = (size_t)b * SS + q0 + hf * 8;
  const int colbase = h * HD + c;
#pragma unroll
  for (int r = 0; r < 8; ++r) {
    const float inv = 1.0f / run_l[r];
    float* op = out + (orow + r) * DD + colbase;
    op[0]  = o0[r] * inv;
    op[16] = o1[r] * inv;
    op[32] = o2[r] * inv;
    op[48] = o3[r] * inv;
  }
}

// ---------------------------------------------------------------------------
extern "C" void kernel_launch(void* const* d_in, const int* in_sizes, int n_in,
                              void* d_out, int out_size, void* d_ws,
                              size_t ws_size, hipStream_t stream) {
  const float* X    = (const float*)d_in[0];  // [B,S,D]
  const float* mask = (const float*)d_in[1];  // [B,1,1,S]
  const float* Wq   = (const float*)d_in[2];
  const float* bq   = (const float*)d_in[3];
  const float* Wk   = (const float*)d_in[4];
  const float* bk   = (const float*)d_in[5];
  const float* Wv   = (const float*)d_in[6];
  const float* bv   = (const float*)d_in[7];
  float* out = (float*)d_out;

  const size_t QKV_ELEMS = (size_t)BB * HH * SS * HD;  // 8,388,608 bf16 each
  unsigned short* ws   = (unsigned short*)d_ws;
  unsigned short* qws  = ws;
  unsigned short* kws  = ws + QKV_ELEMS;
  unsigned short* vtws = ws + 2 * QKV_ELEMS;

  dim3 g1((BB * SS) / 64, DD / 64, 3);           // (128, 16, 3)
  qkv_proj<<<g1, 128, 0, stream>>>(X, Wq, Wk, Wv, bq, bk, bv, qws, kws, vtws);

  const int nwaves = BB * HH * (SS / 16);        // 8192
  flash_attn<<<nwaves / 4, 128, 0, stream>>>(qws, kws, vtws, mask, out);
}
